// MultiTaskGRUAttn_8108898254990
// MI455X (gfx1250) — matrix operand order repaired and checked by reading, not compiled, not run on hardware
//
#include <hip/hip_runtime.h>
#include <hip/hip_bf16.h>

// ---------------------------------------------------------------------------
// MultiTaskGRUAttn for MI455X (gfx1250, wave32, WMMA + async-to-LDS)
//   B=64, T=1024, D=64, H=128  ->  2-layer bi-GRU + attn pooling + heads
// ---------------------------------------------------------------------------

typedef __attribute__((ext_vector_type(16))) _Float16 v16h;
typedef __attribute__((ext_vector_type(8)))  _Float16 v8h;
typedef __attribute__((ext_vector_type(8)))  float    v8f;
typedef __attribute__((ext_vector_type(4)))  float    v4f;
typedef __attribute__((ext_vector_type(4)))  int      v4i;

// Address-space-qualified pointers for the async-to-LDS builtin:
// arg0 = global (AS1) int4*, arg1 = LDS (AS3) int4*.
typedef __attribute__((address_space(1))) v4i* gp_v4i;
typedef __attribute__((address_space(3))) v4i* lp_v4i;

union V16H { v16h v; v8h h[2]; };

constexpr int B_  = 64;
constexpr int T_  = 1024;
constexpr int D_  = 64;
constexpr int H_  = 128;
constexpr int H2  = 2 * H_;   // 256
constexpr int H3  = 3 * H_;   // 384
constexpr int M_  = B_ * T_;  // 65536 rows for the input-projection GEMMs
constexpr int NT_ = H3 / 16;  // 24 N tiles

#define DEV static __device__ __forceinline__

// A fragment: 16x32 f16 tile of row-major [*,K] matrix starting at `base`.
//   lanes 0-15  : M=lane,    halves 0-7 -> K=k0+0..7,   8-15 -> K=k0+16..23
//   lanes 16-31 : M=lane-16, halves 0-7 -> K=k0+8..15,  8-15 -> K=k0+24..31
DEV v16h load_a_frag(const _Float16* base, int K, int k0, int l, int hi) {
  V16H a;
  const _Float16* row = base + (size_t)l * K;
  a.h[0] = *(const v8h*)(row + k0 + hi * 8);
  a.h[1] = *(const v8h*)(row + k0 + 16 + hi * 8);
  return a.v;
}

// B fragment: B[k][n] = W[n0+n][k0+k] with W row-major [N,K]:
//   lanes 0-15: N=lane, K=k0+0..7,16..23; lanes 16-31: N=lane-16, K=k0+8..15,24..31.
DEV v16h load_b_frag(const _Float16* wbase, int K, int k0, int l, int hi) {
  V16H b;
  const _Float16* row = wbase + (size_t)l * K + k0 + hi * 8;
  b.h[0] = *(const v8h*)(row);
  b.h[1] = *(const v8h*)(row + 16);
  return b.v;
}

DEV float sigmoidf_(float x) { return 1.0f / (1.0f + __expf(-x)); }

// ---- async global->LDS copy (16B per lane), guarded fallback -------------
DEV void async_copy_f4(const float* __restrict__ g, float* l) {
#if defined(__gfx1250__) && __has_builtin(__builtin_amdgcn_global_load_async_to_lds_b128)
  __builtin_amdgcn_global_load_async_to_lds_b128(
      (gp_v4i)(uintptr_t)g,
      (lp_v4i)(unsigned)(uintptr_t)l,
      0, 0);
#else
  *(v4f*)l = *(const v4f*)g;
#endif
}

DEV void async_wait_all() {
#if defined(__gfx1250__) && __has_builtin(__builtin_amdgcn_global_load_async_to_lds_b128)
#if __has_builtin(__builtin_amdgcn_s_wait_asynccnt)
  __builtin_amdgcn_s_wait_asynccnt(0);
#else
  asm volatile("s_wait_asynccnt 0" ::: "memory");
#endif
#endif
}

// ---------------------------------------------------------------------------
__global__ void cast_f32_to_f16(const float* __restrict__ src,
                                _Float16* __restrict__ dst, int n) {
  int i = blockIdx.x * blockDim.x + threadIdx.x;
  if (i < n) dst[i] = (_Float16)src[i];
}

// ---------------------------------------------------------------------------
// gi[M,384] = A[M,K] (f16) @ W[384,K]^T (f16) + bias[384], f32 accumulate.
// Each wave computes a 64x16 strip (4 M-tiles) so B fragments are reused 4x.
__global__ void __launch_bounds__(256)
gemm_bias_f16f32(const _Float16* __restrict__ A,
                 const _Float16* __restrict__ W,
                 const float* __restrict__ bias,
                 float* __restrict__ C, int M, int K) {
  int w    = blockIdx.x * (blockDim.x >> 5) + (threadIdx.x >> 5);
  int lane = threadIdx.x & 31;
  int tm = (w / NT_) * 4;  // first of 4 consecutive 16-row tiles
  int tn = w % NT_;
  if (tm * 16 >= M) return;
  int l = lane & 15, hi = lane >> 4;

  const _Float16* wbase = W + (size_t)tn * 16 * K;

  v8f acc[4] = {};
  for (int k0 = 0; k0 < K; k0 += 32) {
    v16h b = load_b_frag(wbase, K, k0, l, hi);
#pragma unroll
    for (int i = 0; i < 4; ++i) {
      v16h a = load_a_frag(A + (size_t)(tm + i) * 16 * K, K, k0, l, hi);
      acc[i] = __builtin_amdgcn_wmma_f32_16x16x32_f16(false, a, false, b,
                                                      (short)0, acc[i],
                                                      false, false);
    }
  }
  int n = tn * 16 + l;
  float bv = bias[n];
#pragma unroll
  for (int i = 0; i < 4; ++i) {
    float* crow = C + ((size_t)((tm + i) * 16 + hi * 8)) * H3 + n;
#pragma unroll
    for (int r = 0; r < 8; ++r) crow[(size_t)r * H3] = acc[i][r] + bv;
  }
}

// ---------------------------------------------------------------------------
// Sequential bi-GRU layer. grid.x = 8: [dir(2) x batch-tile(4)], 256 thr.
// Whh resident in LDS for the whole recurrence; per-step gi tile (24 KB)
// double-buffered in LDS via async global->LDS copies issued one step ahead.
__global__ void __launch_bounds__(256)
gru_seq(const float* __restrict__ gi_f, const float* __restrict__ gi_b,
        const _Float16* __restrict__ whh16_f, const _Float16* __restrict__ whh16_b,
        const float* __restrict__ bhh_f, const float* __restrict__ bhh_b,
        _Float16* __restrict__ out16, float* __restrict__ out32) {
  extern __shared__ char smem[];
  _Float16* sW   = (_Float16*)smem;                        // [384][128] 96 KB
  _Float16* sH16 = (_Float16*)(smem + 98304);              // [16][128]   4 KB
  float*    sGH  = (float*)(smem + 98304 + 4096);          // [16][384]  24 KB
  float*    sHF  = (float*)(smem + 98304 + 4096 + 24576);  // [16][128]   8 KB
  float*    sBH  = (float*)(smem + 135168);                // [384]     1.5 KB
  float*    sGI  = (float*)(smem + 136704);                // 2x[16][384] 48 KB

  const int dir = blockIdx.x >> 2;
  const int bb  = (blockIdx.x & 3) * 16;
  const int tid = threadIdx.x;

  const float*    gi  = dir ? gi_b    : gi_f;
  const _Float16* Wg  = dir ? whh16_b : whh16_f;
  const float*    bhh = dir ? bhh_b   : bhh_f;
  const int coloff    = dir ? H_ : 0;

  // Stage Whh into LDS (16B chunks), bhh, zero h state.
  for (int i = tid; i < (H3 * H_) / 8; i += 256)
    ((v8h*)sW)[i] = ((const v8h*)Wg)[i];
  for (int i = tid; i < H3; i += 256) sBH[i] = bhh[i];
  for (int i = tid; i < 16 * H_; i += 256) {
    sHF[i] = 0.0f;
    sH16[i] = (_Float16)0.0f;
  }

  // Issue gi tile copy for a timestep into one of the two LDS buffers.
  auto issue_gi = [&](int t, int buf) {
#pragma unroll
    for (int i = 0; i < 6; ++i) {
      int idx = tid + i * 256;             // float4 index within 16x384 tile
      int m = idx / 96, c4 = idx % 96;     // 96 float4 per row
      const float* g = gi + (((size_t)(bb + m)) * T_ + t) * H3 + c4 * 4;
      async_copy_f4(g, sGI + (size_t)buf * (16 * H3) + idx * 4);
    }
  };
  issue_gi(dir ? (T_ - 1) : 0, 0);
  async_wait_all();
  __syncthreads();

  const int wv = tid >> 5, lane = tid & 31;
  const int l = lane & 15, hi = lane >> 4;
  const int n0 = wv * 3 * 16;  // 3 N-tiles per wave, 8 waves cover 384

  for (int p = 0; p < T_; ++p) {
    const int t   = dir ? (T_ - 1 - p) : p;
    const int cur = p & 1;

    // Prefetch next step's gi tile into the other buffer (overlaps compute).
    if (p + 1 < T_) issue_gi(dir ? (T_ - 2 - p) : (p + 1), cur ^ 1);

    // gh = h @ Whh^T  (M=16, N=384, K=128)
    v8f acc[3] = {};
#pragma unroll
    for (int k0 = 0; k0 < H_; k0 += 32) {
      v16h a = load_a_frag(sH16, H_, k0, l, hi);
#pragma unroll
      for (int j = 0; j < 3; ++j) {
        v16h b = load_b_frag(sW + (size_t)(n0 + j * 16) * H_, H_, k0, l, hi);
        acc[j] = __builtin_amdgcn_wmma_f32_16x16x32_f16(false, a, false, b,
                                                        (short)0, acc[j],
                                                        false, false);
      }
    }
#pragma unroll
    for (int j = 0; j < 3; ++j) {
      float* dst = sGH + (size_t)(hi * 8) * H3 + (n0 + j * 16 + l);
#pragma unroll
      for (int r = 0; r < 8; ++r) dst[(size_t)r * H3] = acc[j][r];
    }
    __syncthreads();

    // Gate math: 16x128 = 2048 outputs, 8 per thread, gi from LDS.
    const float* gib = sGI + (size_t)cur * (16 * H3);
#pragma unroll
    for (int e = 0; e < 8; ++e) {
      int idx = tid + e * 256;
      int m = idx >> 7, j = idx & 127;
      float i_r = gib[m * H3 + j];
      float i_z = gib[m * H3 + H_ + j];
      float i_n = gib[m * H3 + 2 * H_ + j];
      float g_r = sGH[m * H3 + j] + sBH[j];
      float g_z = sGH[m * H3 + H_ + j] + sBH[H_ + j];
      float g_n = sGH[m * H3 + 2 * H_ + j] + sBH[2 * H_ + j];
      float r  = sigmoidf_(i_r + g_r);
      float z  = sigmoidf_(i_z + g_z);
      float nn = tanhf(i_n + r * g_n);
      float h2 = (1.0f - z) * nn + z * sHF[m * H_ + j];
      sHF[m * H_ + j]  = h2;
      sH16[m * H_ + j] = (_Float16)h2;
      size_t o = (((size_t)(bb + m)) * T_ + t) * H2 + coloff + j;
      if (out16) out16[o] = (_Float16)h2;
      else       out32[o] = h2;
    }
    async_wait_all();   // next-step gi tile landed
    __syncthreads();    // h/gi visible to all waves for next iteration
  }
}

// ---------------------------------------------------------------------------
// Attention pooling + classifier heads. One block per batch sample.
__global__ void __launch_bounds__(256)
attn_heads(const float* __restrict__ z1,           // [B,T,256] f32
           const float* __restrict__ attn_w, const float* __restrict__ attn_b,
           const float* __restrict__ motor_w, const float* __restrict__ motor_b,
           const float* __restrict__ state_w, const float* __restrict__ state_b,
           const int* __restrict__ motor_k, float* __restrict__ out) {
  __shared__ float sS[T_];
  __shared__ float sRed[256];
  __shared__ float sAW[H2];
  __shared__ float sP[H2];

  int b = blockIdx.x, tid = threadIdx.x;
  if (tid < H2) sAW[tid] = attn_w[tid];
  __syncthreads();

  const float* zb = z1 + (size_t)b * T_ * H2;
  for (int t = tid; t < T_; t += 256) {
    const float* zr = zb + (size_t)t * H2;
    float s = 0.0f;
    for (int d = 0; d < H2; ++d) s += zr[d] * sAW[d];
    sS[t] = s + attn_b[0];
  }
  __syncthreads();

  float mx = -INFINITY;
  for (int t = tid; t < T_; t += 256) mx = fmaxf(mx, sS[t]);
  sRed[tid] = mx;
  __syncthreads();
  for (int s = 128; s > 0; s >>= 1) {
    if (tid < s) sRed[tid] = fmaxf(sRed[tid], sRed[tid + s]);
    __syncthreads();
  }
  mx = sRed[0];
  __syncthreads();

  float sm = 0.0f;
  for (int t = tid; t < T_; t += 256) {
    float a = __expf(sS[t] - mx);
    sS[t] = a;
    sm += a;
  }
  sRed[tid] = sm;
  __syncthreads();
  for (int s = 128; s > 0; s >>= 1) {
    if (tid < s) sRed[tid] += sRed[tid + s];
    __syncthreads();
  }
  float inv = 1.0f / sRed[0];
  __syncthreads();

  {  // pooled[d], d = tid in [0,256)
    float p = 0.0f;
    for (int t = 0; t < T_; ++t) p += sS[t] * zb[(size_t)t * H2 + tid];
    sP[tid] = p * inv;
  }
  __syncthreads();

  if (tid < 5) {
    const float* wr = motor_w + (size_t)tid * H2;
    float s = 0.0f;
    for (int d = 0; d < H2; ++d) s += sP[d] * wr[d];
    out[b * 5 + tid] = s + motor_b[tid];
  } else if (tid >= 8 && tid < 10) {
    int cc = tid - 8;
    int k = motor_k[b];
    const float* wr = state_w + ((size_t)k * 2 + cc) * H2;
    float s = 0.0f;
    for (int d = 0; d < H2; ++d) s += sP[d] * wr[d];
    out[B_ * 5 + b * 2 + cc] = s + state_b[k * 2 + cc];
  }
}

// ---------------------------------------------------------------------------
extern "C" void kernel_launch(void* const* d_in, const int* in_sizes, int n_in,
                              void* d_out, int out_size, void* d_ws,
                              size_t ws_size, hipStream_t stream) {
  (void)in_sizes; (void)n_in; (void)out_size; (void)ws_size;

  const float* x       = (const float*)d_in[0];
  const int*   motor_k = (const int*)d_in[1];
  const float* wih[2][2] = {{(const float*)d_in[2],  (const float*)d_in[6]},
                            {(const float*)d_in[10], (const float*)d_in[14]}};
  const float* whh[2][2] = {{(const float*)d_in[3],  (const float*)d_in[7]},
                            {(const float*)d_in[11], (const float*)d_in[15]}};
  const float* bih[2][2] = {{(const float*)d_in[4],  (const float*)d_in[8]},
                            {(const float*)d_in[12], (const float*)d_in[16]}};
  const float* bhh[2][2] = {{(const float*)d_in[5],  (const float*)d_in[9]},
                            {(const float*)d_in[13], (const float*)d_in[17]}};
  const float* attn_w  = (const float*)d_in[18];
  const float* attn_b  = (const float*)d_in[19];
  const float* motor_w = (const float*)d_in[20];
  const float* motor_b = (const float*)d_in[21];
  const float* state_w = (const float*)d_in[22];
  const float* state_b = (const float*)d_in[23];
  float* out = (float*)d_out;

  // ---- workspace carve-up ----
  char* ws = (char*)d_ws;
  size_t off = 0;
  auto alloc = [&](size_t bytes) -> void* {
    void* p = ws + off;
    off = (off + bytes + 255) & ~(size_t)255;
    return p;
  };
  _Float16* x16      = (_Float16*)alloc((size_t)M_ * D_ * 2);
  _Float16* wih16_0f = (_Float16*)alloc((size_t)H3 * D_ * 2);
  _Float16* wih16_0b = (_Float16*)alloc((size_t)H3 * D_ * 2);
  _Float16* wih16_1f = (_Float16*)alloc((size_t)H3 * H2 * 2);
  _Float16* wih16_1b = (_Float16*)alloc((size_t)H3 * H2 * 2);
  _Float16* whh16_0f = (_Float16*)alloc((size_t)H3 * H_ * 2);
  _Float16* whh16_0b = (_Float16*)alloc((size_t)H3 * H_ * 2);
  _Float16* whh16_1f = (_Float16*)alloc((size_t)H3 * H_ * 2);
  _Float16* whh16_1b = (_Float16*)alloc((size_t)H3 * H_ * 2);
  float*    gi0      = (float*)alloc((size_t)M_ * H3 * 4);   // fwd gi
  float*    gi1      = (float*)alloc((size_t)M_ * H3 * 4);   // bwd gi
  _Float16* z0_16    = (_Float16*)alloc((size_t)M_ * H2 * 2);
  float*    z1       = (float*)alloc((size_t)M_ * H2 * 4);

  auto cast = [&](const float* s, _Float16* d, int n) {
    cast_f32_to_f16<<<(n + 255) / 256, 256, 0, stream>>>(s, d, n);
  };

  // ---- f16 staging ----
  cast(x, x16, M_ * D_);
  cast(wih[0][0], wih16_0f, H3 * D_);
  cast(wih[0][1], wih16_0b, H3 * D_);
  cast(wih[1][0], wih16_1f, H3 * H2);
  cast(wih[1][1], wih16_1b, H3 * H2);
  cast(whh[0][0], whh16_0f, H3 * H_);
  cast(whh[0][1], whh16_0b, H3 * H_);
  cast(whh[1][0], whh16_1f, H3 * H_);
  cast(whh[1][1], whh16_1b, H3 * H_);

  const int gemm_blocks = (M_ / 64) * NT_ / 8;  // 3072 (4 M-tiles per wave)
  const size_t gru_smem = 136704 + 49152;       // 185856 B

  // ---- layer 0: input projections + recurrence ----
  gemm_bias_f16f32<<<gemm_blocks, 256, 0, stream>>>(x16, wih16_0f, bih[0][0],
                                                    gi0, M_, D_);
  gemm_bias_f16f32<<<gemm_blocks, 256, 0, stream>>>(x16, wih16_0b, bih[0][1],
                                                    gi1, M_, D_);
  gru_seq<<<8, 256, gru_smem, stream>>>(gi0, gi1, whh16_0f, whh16_0b,
                                        bhh[0][0], bhh[0][1], z0_16, nullptr);

  // ---- layer 1 ----
  gemm_bias_f16f32<<<gemm_blocks, 256, 0, stream>>>(z0_16, wih16_1f, bih[1][0],
                                                    gi0, M_, H2);
  gemm_bias_f16f32<<<gemm_blocks, 256, 0, stream>>>(z0_16, wih16_1b, bih[1][1],
                                                    gi1, M_, H2);
  gru_seq<<<8, 256, gru_smem, stream>>>(gi0, gi1, whh16_1f, whh16_1b,
                                        bhh[1][0], bhh[1][1], nullptr, z1);

  // ---- attention pooling + heads ----
  attn_heads<<<B_, 256, 0, stream>>>(z1, attn_w, attn_b, motor_w, motor_b,
                                     state_w, state_b, motor_k, out);
}
